// NODEAttention_84937273246216
// MI455X (gfx1250) — compile-verified, matrix operand order checked by reading, not run on hardware
//
#include <hip/hip_runtime.h>

typedef unsigned short u16;
typedef __attribute__((ext_vector_type(16))) __bf16 v16bf;
typedef __attribute__((ext_vector_type(8)))  float  v8f;

// ---------- helpers ----------
__device__ __forceinline__ u16 f2bf(float f) {
  union { float f; unsigned u; } x; x.f = f;
  unsigned r = x.u + 0x7fffu + ((x.u >> 16) & 1u);   // round-to-nearest-even
  return (u16)(r >> 16);
}

union BF16Frag { v16bf v; uint4 q[2]; };

// Load one 16-lane-striped A/B fragment half: elements [kb..kb+7] and [kb+16..kb+23]
// base must already point at (row_start + kb).
__device__ __forceinline__ v16bf load_frag(const u16* base) {
  BF16Frag u;
  u.q[0] = *(const uint4*)(base);
  u.q[1] = *(const uint4*)(base + 16);
  return u.v;
}

__device__ __forceinline__ float block_sum(float v, float* red) {
  int tid = threadIdx.x;
  red[tid] = v;
  __syncthreads();
  for (int s = 128; s > 0; s >>= 1) {
    if (tid < s) red[tid] += red[tid + s];
    __syncthreads();
  }
  float r = red[0];
  __syncthreads();
  return r;
}

// CDNA5 async global->LDS copy (ASYNCcnt tracked, no VGPR bounce).
// ldsaddr = LDS byte address (truncated flat addr), goff = 32-bit byte offset, base = SGPR pair.
__device__ __forceinline__ void async_ld_b128(unsigned ldsaddr, unsigned goff, const void* base) {
  asm volatile("global_load_async_to_lds_b128 %0, %1, %2"
               :: "v"(ldsaddr), "v"(goff), "s"(base) : "memory");
}
__device__ __forceinline__ void wait_asynccnt0() {
  asm volatile("s_wait_asynccnt 0" ::: "memory");
}

// ---------- convert kernels ----------
__global__ void cvt_bf16(const float* __restrict__ src, u16* __restrict__ dst, int n) {
  int i = blockIdx.x * 256 + threadIdx.x;
  if (i < n) dst[i] = f2bf(src[i]);
}

// dst[j*K + k] = src[k*Nc + j]  (store weight transposed, bf16)
__global__ void cvt_bf16_T(const float* __restrict__ src, u16* __restrict__ dst, int K, int Nc) {
  int i = blockIdx.x * 256 + threadIdx.x;
  if (i < K * Nc) {
    int k = i / Nc, j = i % Nc;
    dst[(size_t)j * K + k] = f2bf(src[i]);
  }
}

// ---------- WMMA GEMM: C[M,Nc] = A[M,256] * BT[Nc,256]^T + bias ----------
// EPI: 0=QKV scatter, 1=edge KV scatter, 2=FFN1 relu->bf16, 3=FFN2 f32, 4=ROKV f32
template <int EPI>
__global__ __launch_bounds__(256) void gemm16(
    const u16* __restrict__ A, const u16* __restrict__ BT,
    const float* __restrict__ bias,
    u16* __restrict__ ob0, u16* __restrict__ ob1, u16* __restrict__ ob2,
    float* __restrict__ of,
    int M, int tiles_n) {
  const int K = 256;
  int tile  = blockIdx.x * 8 + (threadIdx.x >> 5);
  int total = (M >> 4) * tiles_n;
  if (tile >= total) return;                 // wave-uniform
  int tm = tile / tiles_n, tn = tile % tiles_n;
  int lane = threadIdx.x & 31;
  int lrow = lane & 15;
  int kb0  = (lane < 16) ? 0 : 8;

  const u16* arow = A  + (size_t)(tm * 16 + lrow) * K + kb0;
  const u16* brow = BT + (size_t)(tn * 16 + lrow) * K + kb0;

  v8f c = {0.f, 0.f, 0.f, 0.f, 0.f, 0.f, 0.f, 0.f};
#pragma unroll
  for (int kk = 0; kk < 8; ++kk) {
    if (kk < 7) {
      __builtin_prefetch(arow + (kk + 1) * 32, 0, 1);
      __builtin_prefetch(brow + (kk + 1) * 32, 0, 1);
    }
    v16bf a = load_frag(arow + kk * 32);
    v16bf b = load_frag(brow + kk * 32);
    c = __builtin_amdgcn_wmma_f32_16x16x32_bf16(false, a, false, b, (short)0, c, false, false);
  }

  int j = tn * 16 + lrow;
  float bj = bias ? bias[j] : 0.f;
  int rbase = tm * 16 + ((lane < 16) ? 0 : 8);
#pragma unroll
  for (int r = 0; r < 8; ++r) {
    int m = rbase + r;
    float val = c[r] + bj;
    if (EPI == 0) {                      // node QKV: q bf16 ; k,v into [B,384,256]
      int b_ = m >> 7, n_ = m & 127;
      if (j < 256)      ob0[(size_t)m * 256 + j] = f2bf(val);
      else if (j < 512) ob1[((size_t)b_ * 384 + n_) * 256 + (j - 256)] = f2bf(val);
      else              ob2[((size_t)b_ * 384 + n_) * 256 + (j - 512)] = f2bf(val);
    } else if (EPI == 1) {               // edge KV -> token rows 128..383
      int b_ = m >> 8, e_ = m & 255;
      if (j < 256) ob1[((size_t)b_ * 384 + 128 + e_) * 256 + j]         = f2bf(val);
      else         ob2[((size_t)b_ * 384 + 128 + e_) * 256 + (j - 256)] = f2bf(val);
    } else if (EPI == 2) {               // FFN1: relu -> bf16
      ob0[(size_t)m * 256 + j] = f2bf(val > 0.f ? val : 0.f);
    } else if (EPI == 3) {               // FFN2: f32
      of[(size_t)m * 256 + j] = val;
    } else {                             // ROKV: f32, Nc=512
      of[(size_t)m * 512 + j] = val;
    }
  }
}

// ---------- fused attention per (b,h): 512 blocks, 8 waves, flash-free 1-pass ----------
__global__ __launch_bounds__(256) void attn_kernel(
    const u16* __restrict__ qb,    // [B*128, 256] bf16
    const u16* __restrict__ kbuf,  // [B, 384, 256] bf16
    const u16* __restrict__ vbuf,  // [B, 384, 256] bf16
    const unsigned char* __restrict__ mask,  // [B,128,384]
    float* __restrict__ attn_out)  // [B*128, 256] f32
{
  __shared__ u16 Ksh[384 * 32];    // [token][hd]
  __shared__ u16 Vt[32 * 384];     // [hd][token]  (transposed for B-fragment)
  __shared__ u16 Pb[8 * 16 * 32];  // per-wave P bounce tile

  int b = blockIdx.x >> 3, h = blockIdx.x & 7;
  int tid = threadIdx.x;

  // stage K via CDNA5 async global->LDS (16B chunks, ASYNCcnt tracked)
  for (int cch = tid; cch < 1536; cch += 256) {
    int m = cch >> 2, qq = cch & 3;
    unsigned lds = (unsigned)(uintptr_t)&Ksh[m * 32 + qq * 8];
    unsigned goff = (unsigned)(((( (size_t)b * 384 + m) * 256) + h * 32 + qq * 8) * 2);
    async_ld_b128(lds, goff, kbuf);
  }
  // stage V transposed (manual scatter: async copy cannot transpose)
  for (int i = tid; i < 12288; i += 256) {
    int m = i >> 5, d = i & 31;
    Vt[d * 384 + m] = vbuf[((size_t)b * 384 + m) * 256 + h * 32 + d];
  }
  wait_asynccnt0();
  __syncthreads();

  int wv = tid >> 5, lane = tid & 31;
  int lrow = lane & 15;
  int kb0  = (lane < 16) ? 0 : 8;
  int m0   = wv * 16;
  const float scale = 0.17677669529663687f;  // HD^-0.5

  v16bf aq = load_frag(qb + ((size_t)b * 128 + m0 + lrow) * 256 + h * 32 + kb0);

  v8f o0 = {0.f,0.f,0.f,0.f,0.f,0.f,0.f,0.f};
  v8f o1 = {0.f,0.f,0.f,0.f,0.f,0.f,0.f,0.f};
  v8f rs = {0.f,0.f,0.f,0.f,0.f,0.f,0.f,0.f};
  u16* Pw = &Pb[wv * 16 * 32];
  int rb = (lane < 16) ? 0 : 8;

  for (int ct = 0; ct < 12; ++ct) {
#pragma unroll
    for (int hf = 0; hf < 2; ++hf) {
      int col0 = ct * 32 + hf * 16;
      v16bf bk = load_frag(&Ksh[(col0 + lrow) * 32 + kb0]);
      v8f s = {0.f,0.f,0.f,0.f,0.f,0.f,0.f,0.f};
      s = __builtin_amdgcn_wmma_f32_16x16x32_bf16(false, aq, false, bk, (short)0, s, false, false);
      int col = col0 + lrow;
#pragma unroll
      for (int r = 0; r < 8; ++r) {
        int row = m0 + rb + r;
        unsigned char mk = mask[((size_t)b * 128 + row) * 384 + col];
        float p = mk ? 0.f : __expf(s[r] * scale);
        rs[r] += p;
        Pw[(rb + r) * 32 + hf * 16 + lrow] = f2bf(p);
      }
    }
    asm volatile("s_wait_dscnt 0" ::: "memory");  // our LDS writes -> our LDS reads
    v16bf pa  = load_frag(&Pw[lrow * 32 + kb0]);
    v16bf vb0 = load_frag(&Vt[lrow * 384 + ct * 32 + kb0]);
    v16bf vb1 = load_frag(&Vt[(16 + lrow) * 384 + ct * 32 + kb0]);
    o0 = __builtin_amdgcn_wmma_f32_16x16x32_bf16(false, pa, false, vb0, (short)0, o0, false, false);
    o1 = __builtin_amdgcn_wmma_f32_16x16x32_bf16(false, pa, false, vb1, (short)0, o1, false, false);
  }

#pragma unroll
  for (int r = 0; r < 8; ++r) {
    float v = rs[r];
    v += __shfl_xor(v, 1);
    v += __shfl_xor(v, 2);
    v += __shfl_xor(v, 4);
    v += __shfl_xor(v, 8);
    float inv = 1.0f / v;
    int row = m0 + rb + r;
    float* orow = attn_out + ((size_t)b * 128 + row) * 256 + h * 32;
    orow[lrow]      = o0[r] * inv;
    orow[16 + lrow] = o1[r] * inv;
  }
}

// ---------- residual + LayerNorm ----------
__global__ __launch_bounds__(256) void ln_kernel(
    const float* __restrict__ a, const float* __restrict__ r,
    const float* __restrict__ g, const float* __restrict__ be,
    float* __restrict__ outf, u16* __restrict__ outb, int tok_mode) {
  __shared__ float red[256];
  int row = blockIdx.x, tid = threadIdx.x;
  size_t idx = (size_t)row * 256 + tid;
  float v = a[idx] + r[idx];
  float mean = block_sum(v, red) * (1.0f / 256.0f);
  float d = v - mean;
  float var = block_sum(d * d, red) * (1.0f / 256.0f);
  float y = d * rsqrtf(var + 1e-5f) * g[tid] + be[tid];
  if (outf) outf[idx] = y;
  if (outb) {
    size_t oi;
    if (tok_mode) { int b_ = row >> 7, n_ = row & 127; oi = ((size_t)b_ * 129 + 1 + n_) * 256 + tid; }
    else oi = idx;
    outb[oi] = f2bf(y);
  }
}

__global__ void cls_tok(const float* __restrict__ CLS, u16* __restrict__ tokb) {
  int i = blockIdx.x * 256 + threadIdx.x;
  if (i < 64 * 256) {
    int b_ = i >> 8, j = i & 255;
    tokb[(size_t)b_ * 129 * 256 + j] = f2bf(CLS[i]);
  }
}

// ---------- readout: CLS attention over 129 tokens + LN/FFN/LN (tiny) ----------
__global__ __launch_bounds__(256) void readout_kernel(
    const float* __restrict__ CLS, const unsigned char* __restrict__ cmask,
    const float* __restrict__ rokv,  // [B,129,512] (rk | rv)
    const float* __restrict__ rg1, const float* __restrict__ rbe1,
    const float* __restrict__ rg2, const float* __restrict__ rbe2,
    const float* __restrict__ w1, const float* __restrict__ b1,
    const float* __restrict__ w2, const float* __restrict__ b2,
    float* __restrict__ outc) {
  __shared__ float sc[8 * 129];
  __shared__ float ssum[8];
  __shared__ float c1[256];
  __shared__ float hid[256];
  __shared__ float red[256];
  int b = blockIdx.x, tid = threadIdx.x;
  const float scale = 0.17677669529663687f;

  for (int i = tid; i < 8 * 129; i += 256) {
    int hh = i / 129, m = i % 129;
    const float* kr = rokv + ((size_t)b * 129 + m) * 512 + hh * 32;
    const float* qr = CLS + (size_t)b * 256 + hh * 32;
    float acc = 0.f;
    for (int d = 0; d < 32; ++d) acc += qr[d] * kr[d];
    unsigned char mk = cmask[b * 129 + m];
    sc[i] = mk ? 0.f : __expf(acc * scale);
  }
  __syncthreads();
  if (tid < 8) {
    float s = 0.f;
    for (int m = 0; m < 129; ++m) s += sc[tid * 129 + m];
    ssum[tid] = 1.0f / s;
  }
  __syncthreads();

  int hh = tid >> 5, d = tid & 31;
  float acc = 0.f;
  for (int m = 0; m < 129; ++m)
    acc += sc[hh * 129 + m] * rokv[((size_t)b * 129 + m) * 512 + 256 + hh * 32 + d];
  float v = CLS[(size_t)b * 256 + tid] + acc * ssum[hh];

  float mean = block_sum(v, red) * (1.0f / 256.0f);
  float dd = v - mean;
  float var = block_sum(dd * dd, red) * (1.0f / 256.0f);
  float cv = dd * rsqrtf(var + 1e-5f) * rg1[tid] + rbe1[tid];
  c1[tid] = cv;
  __syncthreads();

  float ha = b1[tid];
  for (int k = 0; k < 256; ++k) ha += c1[k] * w1[(size_t)tid * 256 + k];
  hid[tid] = ha > 0.f ? ha : 0.f;
  __syncthreads();

  float fa = b2[tid];
  for (int k = 0; k < 256; ++k) fa += hid[k] * w2[(size_t)tid * 256 + k];
  float v2 = cv + fa;
  float mean2 = block_sum(v2, red) * (1.0f / 256.0f);
  float d2 = v2 - mean2;
  float var2 = block_sum(d2 * d2, red) * (1.0f / 256.0f);
  outc[(size_t)b * 256 + tid] = d2 * rsqrtf(var2 + 1e-5f) * rg2[tid] + rbe2[tid];
}

// ---------- host launcher ----------
extern "C" void kernel_launch(void* const* d_in, const int* in_sizes, int n_in,
                              void* d_out, int out_size, void* d_ws, size_t ws_size,
                              hipStream_t stream) {
  (void)in_sizes; (void)n_in; (void)out_size; (void)ws_size;
  const float* node_x   = (const float*)d_in[0];
  const float* edge_x   = (const float*)d_in[1];
  const float* CLS      = (const float*)d_in[2];
  const unsigned char* node_mask = (const unsigned char*)d_in[3];
  const unsigned char* CLS_mask  = (const unsigned char*)d_in[4];
  const float* w_qkv    = (const float*)d_in[5];
  const float* b_qkv    = (const float*)d_in[6];
  const float* w_kv_edge= (const float*)d_in[7];
  const float* b_kv_edge= (const float*)d_in[8];
  const float* w1  = (const float*)d_in[9];
  const float* b1  = (const float*)d_in[10];
  const float* w2  = (const float*)d_in[11];
  const float* b2  = (const float*)d_in[12];
  const float* g1  = (const float*)d_in[13];
  const float* be1 = (const float*)d_in[14];
  const float* g2  = (const float*)d_in[15];
  const float* be2 = (const float*)d_in[16];
  const float* ro_w_kv = (const float*)d_in[17];
  const float* ro_b_kv = (const float*)d_in[18];
  const float* ro_w1 = (const float*)d_in[19];
  const float* ro_b1 = (const float*)d_in[20];
  const float* ro_w2 = (const float*)d_in[21];
  const float* ro_b2 = (const float*)d_in[22];
  const float* ro_g1 = (const float*)d_in[23];
  const float* ro_be1= (const float*)d_in[24];
  const float* ro_g2 = (const float*)d_in[25];
  const float* ro_be2= (const float*)d_in[26];

  char* w = (char*)d_ws;
  size_t off = 0;
  auto alloc = [&](size_t bytes) -> void* {
    void* p = w + off;
    off = (off + bytes + 255) & ~(size_t)255;
    return p;
  };

  u16*   nx_b   = (u16*)  alloc(2097152ull * 2);   // node_x bf16
  u16*   ex_b   = (u16*)  alloc(4194304ull * 2);   // edge_x bf16
  u16*   wqkvT  = (u16*)  alloc(196608ull * 2);    // [768,256]
  u16*   wkveT  = (u16*)  alloc(131072ull * 2);    // [512,256]
  u16*   w1b    = (u16*)  alloc(65536ull * 2);     // already [j,k]
  u16*   w2b    = (u16*)  alloc(65536ull * 2);
  u16*   rokvT  = (u16*)  alloc(131072ull * 2);    // [512,256]
  u16*   qb     = (u16*)  alloc(2097152ull * 2);   // [B*128,256]
  u16*   kbv    = (u16*)  alloc(6291456ull * 2);   // [B,384,256]
  u16*   vbv    = (u16*)  alloc(6291456ull * 2);
  float* attn_o = (float*)alloc(2097152ull * 4);
  float* x1f    = (float*)alloc(2097152ull * 4);
  u16*   x1b    = (u16*)  alloc(2097152ull * 2);
  u16*   hb     = (u16*)  alloc(2097152ull * 2);
  float* ff2f   = (float*)alloc(2097152ull * 4);
  u16*   tokb   = (u16*)  alloc(2113536ull * 2);   // [B,129,256]
  float* rokvf  = (float*)alloc(4227072ull * 4);   // [B,129,512]

  // 1) precision conversion (+ weight transposes)
  cvt_bf16<<<8192, 256, 0, stream>>>(node_x, nx_b, 2097152);
  cvt_bf16<<<16384, 256, 0, stream>>>(edge_x, ex_b, 4194304);
  cvt_bf16<<<256, 256, 0, stream>>>(w1, w1b, 65536);
  cvt_bf16<<<256, 256, 0, stream>>>(w2, w2b, 65536);
  cvt_bf16_T<<<768, 256, 0, stream>>>(w_qkv, wqkvT, 256, 768);
  cvt_bf16_T<<<512, 256, 0, stream>>>(w_kv_edge, wkveT, 256, 512);
  cvt_bf16_T<<<512, 256, 0, stream>>>(ro_w_kv, rokvT, 256, 512);

  // 2) projections (WMMA)
  gemm16<0><<<3072, 256, 0, stream>>>(nx_b, wqkvT, b_qkv, qb, kbv, vbv, nullptr, 8192, 48);
  gemm16<1><<<4096, 256, 0, stream>>>(ex_b, wkveT, b_kv_edge, nullptr, kbv, vbv, nullptr, 16384, 32);

  // 3) attention (WMMA, fused softmax, async K staging)
  attn_kernel<<<512, 256, 0, stream>>>(qb, kbv, vbv, node_mask, attn_o);

  // 4) residual+LN, FFN (WMMA), residual+LN -> x (d_out) + tok bf16
  ln_kernel<<<8192, 256, 0, stream>>>(node_x, attn_o, g1, be1, x1f, x1b, 0);
  gemm16<2><<<1024, 256, 0, stream>>>(x1b, w1b, b1, hb, nullptr, nullptr, nullptr, 8192, 16);
  gemm16<3><<<1024, 256, 0, stream>>>(hb, w2b, b2, nullptr, nullptr, nullptr, ff2f, 8192, 16);
  ln_kernel<<<8192, 256, 0, stream>>>(x1f, ff2f, g2, be2, (float*)d_out, tokb, 1);
  cls_tok<<<64, 256, 0, stream>>>(CLS, tokb);

  // 5) readout KV projection (WMMA) + tiny CLS attention / FFN
  gemm16<4><<<2064, 256, 0, stream>>>(tokb, rokvT, ro_b_kv, nullptr, nullptr, nullptr, rokvf, 8256, 32);
  readout_kernel<<<64, 256, 0, stream>>>(CLS, CLS_mask, rokvf, ro_g1, ro_be1, ro_g2, ro_be2,
                                         ro_w1, ro_b1, ro_w2, ro_b2,
                                         (float*)d_out + 2097152);
}
